// UNetTransformerWithAttentionFusion_89455578841777
// MI455X (gfx1250) — matrix operand order repaired and checked by reading, not compiled, not run on hardware
//
#include <hip/hip_runtime.h>

#define BATCH 4
#define NTOK 3136
#define CH 64
#define IMGW 56
#define NT (BATCH*NTOK)       /* 12544 tokens */
#define QTILES (NTOK/16)      /* 196 */
#define SCALE 0.125f          /* d^-0.5, d=64 */

typedef __attribute__((ext_vector_type(16))) __bf16 bf16x16;
typedef __attribute__((ext_vector_type(8)))  float  v8f;
typedef __attribute__((ext_vector_type(4)))  unsigned int u32x4;
typedef unsigned short ushort_t;

union AB { bf16x16 v; u32x4 q[2]; unsigned short h[16]; };

__device__ inline unsigned short f2bf(float f){
  unsigned u = __float_as_uint(f);
  u += 0x7FFFu + ((u >> 16) & 1u);            // round-to-nearest-even
  return (unsigned short)(u >> 16);
}

__device__ inline v8f wmma_bf(AB a, AB b, v8f c){
  return __builtin_amdgcn_wmma_f32_16x16x32_bf16(false, a.v, false, b.v, (short)0, c, false, false);
}

// A-fragment (16x32) from row-major [rows][ld] bf16, at column k0.
// lane<16: row=lane,    K = k0+{0..7, 16..23}
// lane>=16: row=lane-16, K = k0+{8..15,24..31}
__device__ inline AB load_a_rm(const unsigned short* p, int ld, int k0, int lane){
  AB f; int r = lane & 15; int off = (lane >> 4) * 8;
  const unsigned short* s = p + r*ld + k0 + off;
  f.q[0] = *(const u32x4*)s;
  f.q[1] = *(const u32x4*)(s + 16);
  return f;
}

// B-fragment (32x16): element (kk,n) = P[n0+n][k0+kk], P stored n-major with
// contiguous kk (row stride ld). lane<16: n=lane, kk={0..7,16..23}; lane>=16: n=lane-16, kk={8..15,24..31}.
__device__ inline AB load_b_nm(const unsigned short* p, int ld, int n0, int k0, int lane){
  AB f; int r = lane & 15; int off = (lane >> 4) * 8;
  const unsigned short* s = p + (n0 + r)*ld + k0 + off;
  f.q[0] = *(const u32x4*)s;
  f.q[1] = *(const u32x4*)(s + 16);
  return f;
}

// ---------------- weight prep: fp32 -> bf16, transposed where needed ----------------
__global__ void k_wprep(const float* wq, const float* wkv, const float* wproj, const float* w1,
                        ushort_t* wqt, ushort_t* wkvt, ushort_t* wprojt, ushort_t* w1bf){
  int i = blockIdx.x*256 + threadIdx.x;
  if(i < 4096){ int o=i>>6, c=i&63; wqt[o*64+c] = f2bf(wq[c*64+o]); }
  else if(i < 12288){ int j=i-4096;  int o=j>>6, c=j&63; wkvt[o*64+c]   = f2bf(wkv[c*128+o]); }
  else if(i < 16384){ int j=i-12288; int o=j>>6, c=j&63; wprojt[o*64+c] = f2bf(wproj[c*64+o]); }
  else if(i < 49152){ int j=i-16384; w1bf[j] = f2bf(w1[j]); }   // w1 already [co][cin]
}

// ---------------- LN1 + QKV projection (WMMA) ----------------
__global__ void k_ln_qkv(const float* x, const float* g1, const float* b1n,
                         const ushort_t* wqt, const ushort_t* wkvt,
                         const float* bq, const float* bkv,
                         ushort_t* qb, ushort_t* kb, ushort_t* vt){
  __shared__ __align__(16) unsigned short sXn[16][64];
  int blk = blockIdx.x; int b = blk / QTILES; int t = blk % QTILES; int tok0 = t*16;
  int tid = threadIdx.x;
  if(tid < 16){
    const float* xr = x + ((b*NTOK)+tok0+tid)*64;
    float m=0.f; for(int c=0;c<64;++c) m += xr[c]; m *= (1.f/64.f);
    float v=0.f; for(int c=0;c<64;++c){ float d=xr[c]-m; v += d*d; } v *= (1.f/64.f);
    float r = rsqrtf(v + 1e-5f);
    for(int c=0;c<64;++c) sXn[tid][c] = f2bf((xr[c]-m)*r*g1[c] + b1n[c]);
  }
  __syncthreads();
  int lane = tid & 31;
  int wave = __builtin_amdgcn_readfirstlane(tid >> 5);   // provably wave-uniform -> scalar branches
  int col = lane&15, mb = (lane>>4)*8;
  AB a0 = load_a_rm(&sXn[0][0], 64, 0, lane);
  AB a1 = load_a_rm(&sXn[0][0], 64, 32, lane);
  for(int tile = wave*3; tile < wave*3+3; ++tile){   // 12 output col-tiles over 4 waves
    const ushort_t* wsrc; const float* bias; int n0;
    ushort_t* dptr; int dstride;
    if(tile < 4){
      wsrc = wqt;  bias = bq;  n0 = tile*16;
      dptr = qb + ((b*NTOK)+tok0+mb)*64 + n0 + col;  dstride = 64;
    } else if(tile < 8){
      wsrc = wkvt; bias = bkv; n0 = (tile-4)*16;
      dptr = kb + ((b*NTOK)+tok0+mb)*64 + n0 + col;  dstride = 64;
    } else {
      wsrc = wkvt; bias = bkv; n0 = 64 + (tile-8)*16;
      dptr = vt + ((b*64)+(n0-64)+col)*NTOK + tok0 + mb;  dstride = 1;  // V transposed [d][n]
    }
    AB b0  = load_b_nm(wsrc, 64, n0, 0, lane);
    AB b1f = load_b_nm(wsrc, 64, n0, 32, lane);
    v8f c = {}; c = wmma_bf(a0, b0, c); c = wmma_bf(a1, b1f, c);
    float bv = bias[n0 + col];
    #pragma unroll
    for(int j=0;j<8;++j)
      dptr[j*dstride] = f2bf(c[j] + bv);
  }
}

// ---------------- attention: QK^T -> streaming top-8 -> masked softmax -> PV -> proj ----------------
__global__ void k_attn(const ushort_t* qb, const ushort_t* kb, const ushort_t* vt,
                       const ushort_t* wprojt, const float* bproj, float* g){
  __shared__ __align__(16) unsigned short sQ[16][64];
  __shared__ float sS[8][16][16];        // per-wave score staging
  __shared__ float sCand[8][16][8];      // per-wave top-8 candidates
  __shared__ float sKth[16];
  __shared__ float sMax[16];
  __shared__ float sZ[16];
  __shared__ float sO[16][64];
  __shared__ __align__(16) unsigned short sP[8][16][32];   // per-wave exp staging (bf16)
  __shared__ __align__(16) unsigned short sPO[16][64];
  int blk = blockIdx.x, b = blk/QTILES, t = blk%QTILES, q0 = t*16;
  int tid = threadIdx.x, lane = tid & 31;
  int wave = __builtin_amdgcn_readfirstlane(tid >> 5);
  int col = lane&15, mb = (lane>>4)*8;

  for(int i=tid;i<1024;i+=256){
    ((unsigned short*)sQ)[i] = qb[((b*NTOK)+q0)*64 + i];
    ((float*)sO)[i] = 0.f;
  }
  if(tid < 16) sZ[tid] = 0.f;
  __syncthreads();

  AB aq0 = load_a_rm(&sQ[0][0], 64, 0,  lane);
  AB aq1 = load_a_rm(&sQ[0][0], 64, 32, lane);
  const ushort_t* kbb = kb + (b*NTOK)*64;
  const ushort_t* vbb = vt + (b*64)*NTOK;

  // ---- pass 1: scores + streaming per-row top-8 ----
  float top[8];
  #pragma unroll
  for(int j=0;j<8;++j) top[j] = -1e30f;
  for(int ct=wave; ct<QTILES; ct+=8){
    int n0 = ct*16;
    AB b0 = load_b_nm(kbb, 64, n0, 0,  lane);
    AB b1 = load_b_nm(kbb, 64, n0, 32, lane);
    v8f c = {}; c = wmma_bf(aq0, b0, c); c = wmma_bf(aq1, b1, c);
    #pragma unroll
    for(int j=0;j<8;++j) sS[wave][mb+j][col] = c[j]*SCALE;
    asm volatile("s_wait_dscnt 0x0" ::: "memory");
    if(lane < 16){
      #pragma unroll
      for(int cc=0;cc<16;++cc){
        float v = sS[wave][lane][cc];
        if(v > top[7]){ top[7] = v;
          #pragma unroll
          for(int j=7;j>0;--j){ if(top[j]>top[j-1]){ float tp=top[j]; top[j]=top[j-1]; top[j-1]=tp; } }
        }
      }
    }
  }
  if(lane < 16){
    #pragma unroll
    for(int j=0;j<8;++j) sCand[wave][lane][j] = top[j];
  }
  __syncthreads();
  if(tid < 16){                          // merge 8x8 candidates per row -> kth / rowmax
    float t8[8];
    #pragma unroll
    for(int j=0;j<8;++j) t8[j] = -1e30f;
    for(int w2=0;w2<8;++w2)
      for(int j=0;j<8;++j){
        float v = sCand[w2][tid][j];
        if(v > t8[7]){ t8[7] = v;
          for(int jj=7;jj>0;--jj){ if(t8[jj]>t8[jj-1]){ float tp=t8[jj]; t8[jj]=t8[jj-1]; t8[jj-1]=tp; } }
        }
      }
    sKth[tid] = t8[7];
    sMax[tid] = fmaxf(t8[0], 0.f);       // zeros are kept in softmax -> max includes 0
  }
  __syncthreads();

  // ---- pass 2: recompute scores, masked softmax numerator, fused PV ----
  float kth[8], rmx[8], zac[8];
  #pragma unroll
  for(int j=0;j<8;++j){ kth[j]=sKth[mb+j]; rmx[j]=sMax[mb+j]; zac[j]=0.f; }
  v8f oacc[4] = {};
  for(int chk=wave; chk<98; chk+=8){
    int kk0 = chk*32;
    if(chk+8 < 98){
      __builtin_prefetch(kbb + (kk0+256)*64, 0, 1);
      __builtin_prefetch(vbb + kk0 + 256,   0, 1);
    }
    #pragma unroll
    for(int ti=0; ti<2; ++ti){
      int n0 = kk0 + ti*16;
      AB b0 = load_b_nm(kbb, 64, n0, 0,  lane);
      AB b1 = load_b_nm(kbb, 64, n0, 32, lane);
      v8f c = {}; c = wmma_bf(aq0, b0, c); c = wmma_bf(aq1, b1, c);
      #pragma unroll
      for(int j=0;j<8;++j){
        float s = c[j]*SCALE;
        float m = (s >= kth[j]) ? s : 0.f;   // attn*mask: non-topk -> 0, kept in softmax
        float e = __expf(m - rmx[j]);
        zac[j] += e;
        sP[wave][mb+j][ti*16+col] = f2bf(e);
      }
    }
    asm volatile("s_wait_dscnt 0x0" ::: "memory");
    AB ap = load_a_rm(&sP[wave][0][0], 32, 0, lane);
    #pragma unroll
    for(int dt=0; dt<4; ++dt){
      AB bv = load_b_nm(vbb, NTOK, dt*16, kk0, lane);
      oacc[dt] = wmma_bf(ap, bv, oacc[dt]);
    }
  }
  #pragma unroll
  for(int j=0;j<8;++j){                  // reduce Z over 16 columns per half-wave
    float z = zac[j];
    z += __shfl_xor(z, 1, 32); z += __shfl_xor(z, 2, 32);
    z += __shfl_xor(z, 4, 32); z += __shfl_xor(z, 8, 32);
    if(col == 0) atomicAdd(&sZ[mb+j], z);
  }
  #pragma unroll
  for(int dt=0; dt<4; ++dt)
    #pragma unroll
    for(int j=0;j<8;++j)
      atomicAdd(&sO[mb+j][dt*16+col], oacc[dt][j]);
  __syncthreads();

  for(int i=tid;i<1024;i+=256){          // normalize, convert to bf16 for proj GEMM
    int r = i>>6;
    ((unsigned short*)sPO)[i] = f2bf(((float*)sO)[i] / sZ[r]);
  }
  __syncthreads();
  if(wave == 0){                          // fused output projection (16x64)@(64x64)
    AB a0 = load_a_rm(&sPO[0][0], 64, 0,  lane);
    AB a1 = load_a_rm(&sPO[0][0], 64, 32, lane);
    #pragma unroll
    for(int dt=0; dt<4; ++dt){
      int n0 = dt*16;
      AB b0 = load_b_nm(wprojt, 64, n0, 0,  lane);
      AB b1 = load_b_nm(wprojt, 64, n0, 32, lane);
      v8f c = {}; c = wmma_bf(a0, b0, c); c = wmma_bf(a1, b1, c);
      float bv = bproj[n0+col];
      float* gp = g + ((b*NTOK)+q0+mb)*64 + n0 + col;
      #pragma unroll
      for(int j=0;j<8;++j)
        gp[j*64] = c[j] + bv;
    }
  }
}

// ---------------- dense 3x3 local conv + g + LN(norm1) + residual ----------------
__global__ void k_local_res(const float* x, const float* g, const float* lw, const float* lb,
                            const float* g1, const float* b1n, float* x1){
  __shared__ float sv[64];
  int tok = blockIdx.x, co = threadIdx.x;
  int b = tok/NTOK, n = tok%NTOK;
  int yy = n/IMGW, xx = n%IMGW;
  float s = lb[co];
  for(int dy=-1; dy<=1; ++dy){
    int y2 = yy+dy; if((unsigned)y2 >= IMGW) continue;
    for(int dx=-1; dx<=1; ++dx){
      int x2 = xx+dx; if((unsigned)x2 >= IMGW) continue;
      const float* xr = x + ((b*NTOK)+(y2*IMGW+x2))*64;
      const float* wr = lw + (co*64)*9 + (dy+1)*3 + (dx+1);
      for(int ci=0; ci<64; ++ci) s += xr[ci]*wr[ci*9];
    }
  }
  float val = s + g[tok*64+co];
  sv[co] = val;
  __syncthreads();
  float mu=0.f; for(int i=0;i<64;++i) mu += sv[i]; mu *= (1.f/64.f);
  float va=0.f; for(int i=0;i<64;++i){ float d=sv[i]-mu; va += d*d; } va *= (1.f/64.f);
  float r = rsqrtf(va + 1e-5f);
  x1[tok*64+co] = x[tok*64+co] + (val-mu)*r*g1[co] + b1n[co];
}

// ---------------- LN2 -> NCHW image buffer ----------------
__global__ void k_ln2(const float* x1, const float* g2, const float* b2, float* xn2){
  int idx = blockIdx.x*256 + threadIdx.x;
  if(idx >= NT) return;
  int b = idx/NTOK, n = idx%NTOK;
  const float* xr = x1 + idx*64;
  float mu=0.f; for(int c=0;c<64;++c) mu += xr[c]; mu *= (1.f/64.f);
  float va=0.f; for(int c=0;c<64;++c){ float d=xr[c]-mu; va += d*d; } va *= (1.f/64.f);
  float r = rsqrtf(va + 1e-5f);
  for(int c=0;c<64;++c)
    xn2[((b*64)+c)*NTOK + n] = (xr[c]-mu)*r*g2[c] + b2[c];
}

// ---------------- depthwise 3x3 + 5x5 (channel mult 4) -> concat bf16 ----------------
__global__ void k_dw(const float* xn2, const float* w3, const float* b3,
                     const float* w5, const float* b5, ushort_t* cat){
  int idx = blockIdx.x*256 + threadIdx.x;        // B*256*N threads
  int n  = idx % NTOK;
  int oc = (idx / NTOK) & 255;
  int b  = idx / (NTOK*256);
  int ci = oc >> 2;
  const float* img = xn2 + ((b*64)+ci)*NTOK;
  int yy = n/IMGW, xx = n%IMGW;
  float a3 = b3[oc];
  for(int dy=-1; dy<=1; ++dy){ int y2=yy+dy; if((unsigned)y2>=IMGW) continue;
    for(int dx=-1; dx<=1; ++dx){ int x2=xx+dx; if((unsigned)x2>=IMGW) continue;
      a3 += img[y2*IMGW+x2]*w3[oc*9 + (dy+1)*3 + (dx+1)]; } }
  float a5 = b5[oc];
  for(int dy=-2; dy<=2; ++dy){ int y2=yy+dy; if((unsigned)y2>=IMGW) continue;
    for(int dx=-2; dx<=2; ++dx){ int x2=xx+dx; if((unsigned)x2>=IMGW) continue;
      a5 += img[y2*IMGW+x2]*w5[oc*25 + (dy+2)*5 + (dx+2)]; } }
  ushort_t* crow = cat + ((b*NTOK)+n)*512;
  crow[oc]       = f2bf(a3);
  crow[256+oc]   = f2bf(a5);
}

// ---------------- MSFN 1x1 conv as (16x512)@(512x64) WMMA GEMM + residual ----------------
__global__ void k_mlp_out(const ushort_t* cat, const ushort_t* w1bf, const float* b1,
                          const float* x1, float* out){
  int tok0 = blockIdx.x*16;
  int tid = threadIdx.x, lane = tid & 31;
  int wave = __builtin_amdgcn_readfirstlane(tid >> 5);
  int col = lane&15, mb = (lane>>4)*8;
  const ushort_t* crow = cat + tok0*512;
  v8f acc[2] = {};
  for(int k0=0; k0<512; k0+=32){
    AB a = load_a_rm(crow, 512, k0, lane);
    #pragma unroll
    for(int ti=0; ti<2; ++ti){
      AB bb = load_b_nm(w1bf, 512, wave*32 + ti*16, k0, lane);
      acc[ti] = wmma_bf(a, bb, acc[ti]);
    }
  }
  #pragma unroll
  for(int ti=0; ti<2; ++ti){
    int co = wave*32 + ti*16 + col;
    float bv = b1[co];
    const float* xp = x1 + (tok0+mb)*64 + co;
    float* op = out + (tok0+mb)*64 + co;
    #pragma unroll
    for(int j=0;j<8;++j)
      op[j*64] = xp[j*64] + acc[ti][j] + bv;
  }
}

extern "C" void kernel_launch(void* const* d_in, const int* in_sizes, int n_in,
                              void* d_out, int out_size, void* d_ws, size_t ws_size,
                              hipStream_t stream){
  (void)in_sizes; (void)n_in; (void)out_size; (void)ws_size;
  const float* x     = (const float*)d_in[0];
  const float* n1g   = (const float*)d_in[1];
  const float* n1b   = (const float*)d_in[2];
  const float* wq    = (const float*)d_in[3];
  const float* bq    = (const float*)d_in[4];
  const float* wkv   = (const float*)d_in[5];
  const float* bkv   = (const float*)d_in[6];
  const float* wproj = (const float*)d_in[7];
  const float* bproj = (const float*)d_in[8];
  const float* lw    = (const float*)d_in[9];
  const float* lb    = (const float*)d_in[10];
  const float* n2g   = (const float*)d_in[11];
  const float* n2b   = (const float*)d_in[12];
  const float* w3    = (const float*)d_in[13];
  const float* b3    = (const float*)d_in[14];
  const float* w5    = (const float*)d_in[15];
  const float* b5    = (const float*)d_in[16];
  const float* w1    = (const float*)d_in[17];
  const float* b1    = (const float*)d_in[18];
  float* out = (float*)d_out;

  char* ws = (char*)d_ws;
  size_t off = 0;
  auto alloc = [&](size_t bytes)->void*{ void* p = ws + off; off = (off + bytes + 255) & ~(size_t)255; return p; };
  ushort_t* qb     = (ushort_t*)alloc((size_t)NT*64*2);
  ushort_t* kb     = (ushort_t*)alloc((size_t)NT*64*2);
  ushort_t* vt     = (ushort_t*)alloc((size_t)NT*64*2);
  float*    g      = (float*)   alloc((size_t)NT*64*4);
  float*    x1     = (float*)   alloc((size_t)NT*64*4);
  float*    xn2    = (float*)   alloc((size_t)NT*64*4);
  ushort_t* cat    = (ushort_t*)alloc((size_t)NT*512*2);
  ushort_t* wqt    = (ushort_t*)alloc(4096*2);
  ushort_t* wkvt   = (ushort_t*)alloc(8192*2);
  ushort_t* wprojt = (ushort_t*)alloc(4096*2);
  ushort_t* w1bf   = (ushort_t*)alloc(32768*2);

  k_wprep   <<<192,            256, 0, stream>>>(wq, wkv, wproj, w1, wqt, wkvt, wprojt, w1bf);
  k_ln_qkv  <<<BATCH*QTILES,   128, 0, stream>>>(x, n1g, n1b, wqt, wkvt, bq, bkv, qb, kb, vt);
  k_attn    <<<BATCH*QTILES,   256, 0, stream>>>(qb, kb, vt, wprojt, bproj, g);
  k_local_res<<<NT,             64, 0, stream>>>(x, g, lw, lb, n1g, n1b, x1);
  k_ln2     <<<(NT+255)/256,   256, 0, stream>>>(x1, n2g, n2b, xn2);
  k_dw      <<<NT,             256, 0, stream>>>(xn2, w3, b3, w5, b5, cat);
  k_mlp_out <<<NT/16,           64, 0, stream>>>(cat, w1bf, b1, x1, out);
}
